// FUDFT_73340861547123
// MI455X (gfx1250) — matrix operand order repaired and checked by reading, not compiled, not run on hardware
//
#include <hip/hip_runtime.h>
#include <math.h>

// FUDFT on MI455X (gfx1250, wave32):
//   out[b,s,c] = | sum_n exp(i*2pi*w[s]*tau[b,n]) * x[b,n,c] | / sqrt(N), rows 0/S-1 zeroed
// Two f16 WMMA GEMMs (cos-plane / sin-plane) sharing the B operand, f32 accum.
// v2: wave grid 4(m) x 2(n) -> 1 trig pair per WMMA (was 2); angles kept in
// revolutions so raw v_sin_f32/v_cos_f32 need no 1/(2pi) pre-mul; staging uses
// v_cvt_pk_f16_f32 + ds_store_b32 on (k,k+1) pairs to halve convert/store count.

typedef __attribute__((ext_vector_type(16))) _Float16 v16h;
typedef __attribute__((ext_vector_type(8)))  _Float16 v8h;
typedef __attribute__((ext_vector_type(2)))  __fp16   v2hf;  // cvt_pkrtz result type
typedef __attribute__((ext_vector_type(8)))  float    v8f;
typedef __attribute__((ext_vector_type(4)))  float    v4f;

#define NB    8
#define NN    2048   // signal length (K dimension)
#define NS    2048   // frequency bins (M dimension)
#define NC    256    // channels (N dimension)
#define CHUNK 32     // K-step per iteration (matches WMMA K=32)
#define PITCH 40     // halves per LDS column: 80B = 16B-aligned, bank stride 20 -> conflict-free

__global__ __launch_bounds__(256)
void fudft_wmma_kernel(const float* __restrict__ x,
                       const float* __restrict__ t,
                       const float* __restrict__ freqs,
                       float* __restrict__ out)
{
    __shared__ float tauS[NN];                               // 8 KB: t[b,:] * (N-1)
    __shared__ __align__(16) _Float16 Bbuf[2][NC * PITCH];   // 2 x 20 KB double buffer

    const int tid   = threadIdx.x;
    const int lane  = tid & 31;
    const int wave  = tid >> 5;
    const int wm    = wave >> 1;          // 0..3 : wave row (16 s each)
    const int wn    = wave & 1;           // 0..1 : wave col (128 c each)
    const int stile = blockIdx.x;         // 0..NS/64-1
    const int b     = blockIdx.y;         // 0..NB-1

    const float* xb = x + (size_t)b * NN * NC;
    const float* tb = t + (size_t)b * NN;

    // ---- stage tau = t*(N-1) into LDS (whole row, once) ----
    for (int i = tid; i < NN; i += 256)
        tauS[i] = tb[i] * (float)(NN - 1);

    // ---- per-lane angular rate, in REVOLUTIONS: theta_rev = wrev * tau ----
    // rows[s] = -2*pi * freqs[s]*(N-1)/N  ->  wrev = -freqs[s]*(N-1)/N
    // (sign of sin irrelevant for |y|; hardware v_sin/v_cos take revolutions)
    const int   sBase = stile * 64 + wm * 16;
    const int   row   = lane & 15;        // A/D layout: both lane halves map rows 0..15
    const float wrev  = -freqs[sBase + row] * ((float)(NN - 1) / (float)NN);

    // ---- accumulators: 8 n-subtiles x {re,im} ----
    v8f accRe[8], accIm[8];
    #pragma unroll
    for (int j = 0; j < 8; ++j)
        #pragma unroll
        for (int e = 0; e < 8; ++e) { accRe[j][e] = 0.0f; accIm[j][e] = 0.0f; }

    // ---- stage one 32 x 256 chunk of x into LDS as f16, column-major (pitch 40) ----
    // Each thread owns 4 columns and 8 k-rows as (k,k+1) pairs so converts are
    // v_cvt_pk_f16_f32 and LDS stores are b32 (pairs are k-contiguous in a column).
    auto stage = [&](int kc, int bi) {
        const float* src = xb + (size_t)(kc * CHUNK) * NC;
        if (kc + 1 < NN / CHUNK)  // speculative prefetch of the chunk after next
            __builtin_prefetch(src + (size_t)CHUNK * NC, 0, 0);
        const int c4 = (tid & 63) * 4;    // 64 threads span the 256 columns (float4)
        const int kp = (tid >> 6) * 2;    // 4 k-phases of row pairs
        #pragma unroll
        for (int it = 0; it < 4; ++it) {
            const int k = kp + it * 8;    // even rows 0..30
            v4f v0 = *(const v4f*)(src + (size_t)k       * NC + c4);
            v4f v1 = *(const v4f*)(src + (size_t)(k + 1) * NC + c4);
            #pragma unroll
            for (int j = 0; j < 4; ++j) {
                v2hf p = __builtin_amdgcn_cvt_pkrtz(v0[j], v1[j]);   // (k, k+1)
                *(v2hf*)&Bbuf[bi][(c4 + j) * PITCH + k] = p;          // 4B-aligned
            }
        }
    };

    const int kchunks = NN / CHUNK;       // 64
    stage(0, 0);
    __syncthreads();

    for (int kc = 0; kc < kchunks; ++kc) {
        const int bi = kc & 1;
        if (kc + 1 < kchunks) stage(kc + 1, bi ^ 1);

        // ---- tau values for this lane's A-fragment slots ----
        // A 16x32 f16, contiguous k order chosen to match B (not the ISA table order): lane group base = (lane/16)*16; halves 0..15 -> K base+0..15
        // (element e of v16h = VGPR e/2, half e&1)
        const int kA = kc * CHUNK + ((lane >> 4) << 4);
        float tau16[16];
        {
            const v4f* tp = (const v4f*)&tauS[kA];        // 32B-aligned (kA % 8 == 0)
            v4f a0 = tp[0], a1 = tp[1], b0 = tp[2], b1 = tp[3];
            #pragma unroll
            for (int j = 0; j < 4; ++j) {
                tau16[j]      = a0[j]; tau16[4 + j]  = a1[j];
                tau16[8 + j]  = b0[j]; tau16[12 + j] = b1[j];
            }
        }

        // ---- B fragments: lane = column (lane&15), K rows (lane/16)*16 .. +15 ----
        v16h Bf[8];
        const int kB = (lane >> 4) << 4;
        #pragma unroll
        for (int nn = 0; nn < 8; ++nn) {
            const int cfr = wn * 128 + nn * 16 + (lane & 15);
            const v8h* p = (const v8h*)&Bbuf[bi][cfr * PITCH + kB];  // 16B aligned
            v8h lo = p[0], hi = p[1];
            #pragma unroll
            for (int e = 0; e < 8; ++e) { Bf[nn][e] = lo[e]; Bf[nn][8 + e] = hi[e]; }
        }

        // ---- synthesize one trig A-fragment pair, feed 16 WMMAs ----
        v16h Ac, As;
        #pragma unroll
        for (int e = 0; e < 16; ++e) {
            const float rev = wrev * tau16[e];               // angle in revolutions
            Ac[e] = (_Float16)__builtin_amdgcn_cosf(rev);    // raw v_cos_f32
            As[e] = (_Float16)__builtin_amdgcn_sinf(rev);    // raw v_sin_f32
        }
        #pragma unroll
        for (int nn = 0; nn < 8; ++nn) {
            accRe[nn] = __builtin_amdgcn_wmma_f32_16x16x32_f16(
                false, Ac, false, Bf[nn], (short)0, accRe[nn], false, false);
            accIm[nn] = __builtin_amdgcn_wmma_f32_16x16x32_f16(
                false, As, false, Bf[nn], (short)0, accIm[nn], false, false);
        }
        __syncthreads();
    }

    // ---- epilogue: magnitude / sqrt(N), zero rows 0 and S-1, coalesced stores ----
    const float scale = 1.0f / sqrtf((float)NN);
    const int   mrow  = (lane >> 4) << 3;   // D layout: lanes 16-31 hold M = r+8
    const int   ccol  = wn * 128 + (lane & 15);
    #pragma unroll
    for (int nn = 0; nn < 8; ++nn) {
        #pragma unroll
        for (int r = 0; r < 8; ++r) {
            const int s = sBase + r + mrow;
            const int c = ccol + nn * 16;
            const float re = accRe[nn][r];
            const float im = accIm[nn][r];
            float m = __fsqrt_rn(re * re + im * im) * scale;
            if (s == 0 || s == NS - 1) m = 0.0f;
            out[((size_t)b * NS + s) * NC + c] = m;
        }
    }
}

extern "C" void kernel_launch(void* const* d_in, const int* in_sizes, int n_in,
                              void* d_out, int out_size, void* d_ws, size_t ws_size,
                              hipStream_t stream) {
    const float* x     = (const float*)d_in[0];   // (B, N, C) f32
    const float* t     = (const float*)d_in[1];   // (B, N)    f32
    const float* freqs = (const float*)d_in[2];   // (S,)      f32
    (void)in_sizes; (void)n_in; (void)out_size; (void)d_ws; (void)ws_size;

    dim3 grid(NS / 64, NB);   // 32 s-tiles x 8 batches = 256 workgroups
    fudft_wmma_kernel<<<grid, 256, 0, stream>>>(x, t, freqs, (float*)d_out);
}
